// ParticleNetLaplace_60722247630941
// MI455X (gfx1250) — matrix operand order repaired and checked by reading, not compiled, not run on hardware
//
#include <hip/hip_runtime.h>
#include <hip/hip_bf16.h>

typedef __attribute__((ext_vector_type(16))) _Float16 v16h;
typedef __attribute__((ext_vector_type(8)))  _Float16 v8h;
typedef __attribute__((ext_vector_type(8)))  float    v8f;

#define PN_EPS 1e-3f
#define PN_B   64
#define PN_N   256
#define PN_K   16
// edge columns per block: col = (b*N + n)*K + k
#define PN_NCOLS ((long)PN_B * PN_N * PN_K)   // 262144

// ---------------------------------------------------------------------------
// 0. global mean / inv-std over the whole input tensor (single block)
// ---------------------------------------------------------------------------
__global__ void pn_gstats(const float* __restrict__ X, long n, float* __restrict__ gs) {
    __shared__ float r1[256], r2[256];
    float s = 0.f, s2 = 0.f;
    for (long i = threadIdx.x; i < n; i += 256) {
        float v = X[i]; s += v; s2 += v * v;
    }
    r1[threadIdx.x] = s; r2[threadIdx.x] = s2;
    __syncthreads();
    for (int st = 128; st > 0; st >>= 1) {
        if ((int)threadIdx.x < st) { r1[threadIdx.x] += r1[threadIdx.x + st]; r2[threadIdx.x] += r2[threadIdx.x + st]; }
        __syncthreads();
    }
    if (threadIdx.x == 0) {
        float mean = r1[0] / (float)n;
        float var  = r2[0] / (float)n - mean * mean;
        gs[0] = mean;
        gs[1] = rsqrtf(var + PN_EPS);
    }
}

__global__ void pn_ftrans(const float* __restrict__ X, float* __restrict__ Xn,
                          const float* __restrict__ gs,
                          const float* __restrict__ gamma, const float* __restrict__ beta, long n) {
    long i = (long)blockIdx.x * blockDim.x + threadIdx.x;
    if (i >= n) return;
    Xn[i] = (X[i] - gs[0]) * gs[1] * gamma[0] + beta[0];
}

// ---------------------------------------------------------------------------
// 1. kNN on first 6 channels. One workgroup (256 threads) per batch element.
// ---------------------------------------------------------------------------
__global__ void pn_knn(const float* __restrict__ Xin, int Cin, int* __restrict__ nnidx) {
    __shared__ float pts[6 * PN_N];
    const int b = blockIdx.x, t = threadIdx.x;
    for (int c = 0; c < 6; ++c)
        pts[c * PN_N + t] = Xin[((size_t)b * Cin + c) * PN_N + t];
    __syncthreads();

    float p[6];
#pragma unroll
    for (int c = 0; c < 6; ++c) p[c] = pts[c * PN_N + t];

    float bd[PN_K]; int bi[PN_K];
#pragma unroll
    for (int k = 0; k < PN_K; ++k) { bd[k] = 3.0e38f; bi[k] = 0; }

    for (int j = 0; j < PN_N; ++j) {
        if (j == t) continue;
        float d = 0.f;
#pragma unroll
        for (int c = 0; c < 6; ++c) { float df = pts[c * PN_N + j] - p[c]; d += df * df; }
        if (d < bd[PN_K - 1]) {
            float cd = d; int ci = j;
#pragma unroll
            for (int k = 0; k < PN_K; ++k) {
                if (cd < bd[k]) {
                    float td = bd[k]; int ti = bi[k];
                    bd[k] = cd; bi[k] = ci; cd = td; ci = ti;
                }
            }
        }
    }
#pragma unroll
    for (int k = 0; k < PN_K; ++k)
        nnidx[((size_t)b * PN_N + t) * PN_K + k] = bi[k];
}

// ---------------------------------------------------------------------------
// 2. build edge features, EDGE-MAJOR: H[col][Cpad] = [x_nn - x_c ; x_c ; 0pad]
// ---------------------------------------------------------------------------
__global__ void pn_build_h(const float* __restrict__ Xin, const int* __restrict__ nnidx,
                           _Float16* __restrict__ H, int Cin, int Cpad) {
    long col = (long)blockIdx.x * blockDim.x + threadIdx.x;
    if (col >= PN_NCOLS) return;
    long bn = col >> 4;                    // b*N + n
    int  n  = (int)(bn & (PN_N - 1));
    int  b  = (int)(bn >> 8);
    int  nn = nnidx[col];
    const float* Xb = Xin + (size_t)b * Cin * PN_N;
    _Float16* Hrow = H + (size_t)col * Cpad;
    for (int c = 0; c < Cin; ++c) {
        float xc = Xb[c * PN_N + n];
        float xn = Xb[c * PN_N + nn];
        Hrow[c]       = (_Float16)(xn - xc);
        Hrow[Cin + c] = (_Float16)xc;
    }
    for (int c = 2 * Cin; c < Cpad; ++c) Hrow[c] = (_Float16)0.f;
}

// ---------------------------------------------------------------------------
// 3. weight convert fp32 -> f16, PRE-SWIZZLED into A-fragment order:
//    WhS[(kb/32)*O + o][j] , j = hi*16 + e , channel = kb + j
// ---------------------------------------------------------------------------
__global__ void pn_convw(const float* __restrict__ W, _Float16* __restrict__ WhS,
                         int O, int CinEff, int Cpad) {
    int i = blockIdx.x * blockDim.x + threadIdx.x;
    if (i >= O * Cpad) return;
    int kb5 = i / (O * 32);
    int rem = i - kb5 * O * 32;
    int o = rem >> 5;
    int j = rem & 31;
    int hi = j >> 4, e = j & 15;
    int c = kb5 * 32 + j;
    WhS[i] = (c < CinEff) ? (_Float16)W[o * CinEff + c] : (_Float16)0.f;
}

// ---------------------------------------------------------------------------
// 4. WMMA GEMM: D[col][o] = relu(sum_c W[o][c]*H[col][c] + bias[o])
//    One workgroup = O/16 waves, 64 columns. H tile staged to LDS via
//    global_load_async_to_lds_b128 (ASYNCcnt), fragments via ds_load_b128.
// ---------------------------------------------------------------------------
template<int CP, int OC>
__global__ __launch_bounds__((OC / 16) * 32)
void pn_gemm(const _Float16* __restrict__ WhS, const _Float16* __restrict__ H,
             _Float16* __restrict__ D, const float* __restrict__ bias) {
    constexpr int STR = CP + 16;                 // padded LDS row (halfs), 32B-aligned
    constexpr int NT  = (OC / 16) * 32;
    constexpr int CHUNKS = CP / 8;               // 16-byte chunks per column
    constexpr int TOTAL  = 64 * CHUNKS;
    __shared__ _Float16 sh[64 * STR];

    const int  tid  = threadIdx.x;
    const long col0 = (long)blockIdx.x * 64;
    const _Float16* src = H + (size_t)col0 * CP;

    // ---- async stage: 64 cols x CP halfs, global -> LDS ----
    for (int v = tid; v < TOTAL; v += NT) {
        int ccol = v / CHUNKS, j = v - ccol * CHUNKS;
        unsigned lds = (unsigned)(uintptr_t)(&sh[ccol * STR + j * 8]);
        unsigned long long g = (unsigned long long)(uintptr_t)(src + (size_t)v * 8);
        asm volatile("global_load_async_to_lds_b128 %0, %1, off"
                     :: "v"(lds), "v"(g) : "memory");
    }
    __builtin_prefetch(src + (size_t)64 * CP, 0, 1);   // next workgroup's tile
    asm volatile("s_wait_asynccnt 0x0" ::: "memory");
    __syncthreads();

    const int lane = tid & 31, wv = tid >> 5;
    const int hi = lane >> 4, ln = lane & 15;
    const int row0 = wv * 16;

    v8f acc[4] = {};
#pragma unroll
    for (int kb = 0; kb < CP; kb += 32) {
        const v16h a = *(const v16h*)(WhS + ((size_t)(kb >> 5) * OC + row0 + ln) * 32 + hi * 16);
#pragma unroll
        for (int nt = 0; nt < 4; ++nt) {
            const v16h bf = *(const v16h*)(sh + (nt * 16 + ln) * STR + kb + hi * 16);
            acc[nt] = __builtin_amdgcn_wmma_f32_16x16x32_f16(
                false, a, false, bf, (short)0, acc[nt], false, false);
        }
    }
#pragma unroll
    for (int nt = 0; nt < 4; ++nt) {
        v8h o8;
#pragma unroll
        for (int r = 0; r < 8; ++r) {
            float v = acc[nt][r] + bias[row0 + hi * 8 + r];
            o8[r] = (_Float16)fmaxf(v, 0.f);
        }
        *(v8h*)(D + (size_t)(col0 + nt * 16 + ln) * OC + row0 + hi * 8) = o8;
    }
}

// ---------------------------------------------------------------------------
// 5. per-channel sum/sumsq over edge-major f16 activations D[col][O]
//    (channel fixed per thread -> register accumulate, LDS then global atomics)
// ---------------------------------------------------------------------------
__global__ void pn_chstats_h(const _Float16* __restrict__ D, long ncols, int O,
                             float* __restrict__ sum, float* __restrict__ sq) {
    __shared__ float s1[128], s2[128];
    for (int i = threadIdx.x; i < O; i += blockDim.x) { s1[i] = 0.f; s2[i] = 0.f; }
    __syncthreads();
    const int c   = threadIdx.x & (O - 1);
    const int grp = threadIdx.x / O;
    const int gpb = blockDim.x / O;
    float s = 0.f, s2r = 0.f;
    for (long col = (long)blockIdx.x * gpb + grp; col < ncols; col += (long)gridDim.x * gpb) {
        float v = (float)D[col * O + c];
        s += v; s2r += v * v;
    }
    atomicAdd(&s1[c], s);
    atomicAdd(&s2[c], s2r);
    __syncthreads();
    for (int i = threadIdx.x; i < O; i += blockDim.x) {
        atomicAdd(&sum[i], s1[i]);
        atomicAdd(&sq[i],  s2[i]);
    }
}

// per-channel stats over channel-major fp32 buffer [O][ncols] (shortcut path)
__global__ void pn_chstats_f(const float* __restrict__ Dbuf, long ncols,
                             float* __restrict__ sum, float* __restrict__ sq) {
    const int o = blockIdx.y;
    const float* row = Dbuf + (size_t)o * ncols;
    float s = 0.f, s2 = 0.f;
    for (long i = (long)blockIdx.x * blockDim.x + threadIdx.x; i < ncols;
         i += (long)gridDim.x * blockDim.x) {
        float v = row[i]; s += v; s2 += v * v;
    }
    __shared__ float r1[256], r2[256];
    r1[threadIdx.x] = s; r2[threadIdx.x] = s2;
    __syncthreads();
    for (int st = 128; st > 0; st >>= 1) {
        if ((int)threadIdx.x < st) { r1[threadIdx.x] += r1[threadIdx.x + st]; r2[threadIdx.x] += r2[threadIdx.x + st]; }
        __syncthreads();
    }
    if (threadIdx.x == 0) { atomicAdd(&sum[o], r1[0]); atomicAdd(&sq[o], r2[0]); }
}

// ---------------------------------------------------------------------------
// 6. batch-norm apply, in place on edge-major f16 buffer (channel = i & (O-1))
// ---------------------------------------------------------------------------
__global__ void pn_bn_apply(_Float16* __restrict__ Dbuf, int O, float cnt,
                            const float* __restrict__ sum, const float* __restrict__ sq,
                            const float* __restrict__ g, const float* __restrict__ bt,
                            long total) {
    long i = (long)blockIdx.x * blockDim.x + threadIdx.x;
    if (i >= total) return;
    int c = (int)(i & (O - 1));
    float m   = sum[c] / cnt;
    float var = sq[c] / cnt - m * m;
    float is  = rsqrtf(var + PN_EPS);
    float v   = (float)Dbuf[i];
    Dbuf[i]   = (_Float16)((v - m) * is * g[c] + bt[c]);
}

// ---------------------------------------------------------------------------
// 7. shortcut GEMM (small): sc[o][b*N+n] = sum_c scW[o][c]*Xin[b][c][n]
// ---------------------------------------------------------------------------
__global__ void pn_scgemm(const float* __restrict__ scW, const float* __restrict__ Xin,
                          float* __restrict__ sc, int O, int Cin) {
    long i = (long)blockIdx.x * blockDim.x + threadIdx.x;
    long total = (long)O * PN_B * PN_N;
    if (i >= total) return;
    int o  = (int)(i / (PN_B * PN_N));
    long bn = i % (PN_B * PN_N);
    int b  = (int)(bn >> 8);
    int n  = (int)(bn & (PN_N - 1));
    float acc = 0.f;
    for (int c = 0; c < Cin; ++c)
        acc += scW[o * Cin + c] * Xin[((size_t)b * Cin + c) * PN_N + n];
    sc[i] = acc;
}

// ---------------------------------------------------------------------------
// 8. combine: Xout[b][o][n] = relu(mean_k D3[col][o] + BN(sc))
// ---------------------------------------------------------------------------
__global__ void pn_combine(const _Float16* __restrict__ D3, const float* __restrict__ sc,
                           const float* __restrict__ ssum, const float* __restrict__ ssq,
                           const float* __restrict__ g, const float* __restrict__ bt,
                           float* __restrict__ Xout, int O) {
    long i = (long)blockIdx.x * blockDim.x + threadIdx.x;
    long total = (long)PN_B * O * PN_N;
    if (i >= total) return;
    int n = (int)(i & (PN_N - 1));
    long t = i >> 8;
    int o = (int)(t % O);
    int b = (int)(t / O);
    long colbase = ((long)b * PN_N + n) * PN_K;
    float p = 0.f;
#pragma unroll
    for (int k = 0; k < PN_K; ++k)
        p += (float)D3[(size_t)(colbase + k) * O + o];
    p *= (1.f / PN_K);
    const float cnt = (float)(PN_B * PN_N);
    float m   = ssum[o] / cnt;
    float var = ssq[o] / cnt - m * m;
    float s   = sc[(size_t)o * PN_B * PN_N + (size_t)b * PN_N + n];
    float v   = p + (s - m) * rsqrtf(var + PN_EPS) * g[o] + bt[o];
    Xout[((size_t)b * O + o) * PN_N + n] = fmaxf(v, 0.f);
}

// ---------------------------------------------------------------------------
// 9. final mean over N -> out[b][o]
// ---------------------------------------------------------------------------
__global__ void pn_finalmean(const float* __restrict__ X3, float* __restrict__ out, int O) {
    int i = blockIdx.x * blockDim.x + threadIdx.x;
    if (i >= PN_B * O) return;
    int b = i / O, o = i % O;
    const float* row = X3 + ((size_t)b * O + o) * PN_N;
    float s = 0.f;
    for (int n = 0; n < PN_N; ++n) s += row[n];
    out[i] = s * (1.f / PN_N);
}

// ---------------------------------------------------------------------------
// host orchestration
// ---------------------------------------------------------------------------
struct PnLayer { const float *W, *b, *g, *bt; int CinEff, Cpad; };

static void pn_run_block(const float* Xin, int Cin, int O,
                         const PnLayer layers[3],
                         const float* scW, const float* scG, const float* scB,
                         float* Xout,
                         int* nnidx, _Float16* bufA, _Float16* bufB, _Float16* Wh,
                         float* chsum, float* chsq, float* scsum, float* scsq, float* scbuf,
                         hipStream_t stream) {
    const long NC = PN_NCOLS;
    pn_knn<<<PN_B, PN_N, 0, stream>>>(Xin, Cin, nnidx);
    pn_build_h<<<(int)((NC + 255) / 256), 256, 0, stream>>>(Xin, nnidx, bufA, Cin, layers[0].Cpad);

    _Float16* Hcur = bufA;
    _Float16* Dcur = bufB;
    for (int l = 0; l < 3; ++l) {
        const int CP = layers[l].Cpad;
        pn_convw<<<(O * CP + 255) / 256, 256, 0, stream>>>(
            layers[l].W, Wh, O, layers[l].CinEff, CP);
        hipMemsetAsync(chsum, 0, 1024, stream);            // zeroes chsum + chsq
        const int nblk = (int)(NC / 64);
        if (CP == 32)
            pn_gemm<32, 64><<<nblk, 128, 0, stream>>>(Wh, Hcur, Dcur, layers[l].b);
        else if (CP == 64)
            pn_gemm<64, 64><<<nblk, 128, 0, stream>>>(Wh, Hcur, Dcur, layers[l].b);
        else
            pn_gemm<128, 128><<<nblk, 256, 0, stream>>>(Wh, Hcur, Dcur, layers[l].b);
        pn_chstats_h<<<256, 256, 0, stream>>>(Dcur, NC, O, chsum, chsq);
        long total = (long)O * NC;
        pn_bn_apply<<<(int)((total + 255) / 256), 256, 0, stream>>>(
            Dcur, O, (float)NC, chsum, chsq, layers[l].g, layers[l].bt, total);
        _Float16* tmp = Hcur; Hcur = Dcur; Dcur = tmp;
    }
    // shortcut path
    hipMemsetAsync(scsum, 0, 1024, stream);                // zeroes scsum + scsq
    long scTot = (long)O * PN_B * PN_N;
    pn_scgemm<<<(int)((scTot + 255) / 256), 256, 0, stream>>>(scW, Xin, scbuf, O, Cin);
    pn_chstats_f<<<dim3(16, O), 256, 0, stream>>>(scbuf, (long)PN_B * PN_N, scsum, scsq);
    long cTot = (long)PN_B * O * PN_N;
    pn_combine<<<(int)((cTot + 255) / 256), 256, 0, stream>>>(
        Hcur, scbuf, scsum, scsq, scG, scB, Xout, O);
}

extern "C" void kernel_launch(void* const* d_in, const int* in_sizes, int n_in,
                              void* d_out, int out_size, void* d_ws, size_t ws_size,
                              hipStream_t stream) {
    (void)in_sizes; (void)n_in; (void)out_size; (void)ws_size;
    const float* X    = (const float*)d_in[0];
    const float* ftg  = (const float*)d_in[1];
    const float* ftb  = (const float*)d_in[2];

    PnLayer l1[3], l2[3];
    for (int l = 0; l < 3; ++l) {
        l1[l].W  = (const float*)d_in[3 + 4 * l];
        l1[l].b  = (const float*)d_in[4 + 4 * l];
        l1[l].g  = (const float*)d_in[5 + 4 * l];
        l1[l].bt = (const float*)d_in[6 + 4 * l];
        l2[l].W  = (const float*)d_in[18 + 4 * l];
        l2[l].b  = (const float*)d_in[19 + 4 * l];
        l2[l].g  = (const float*)d_in[20 + 4 * l];
        l2[l].bt = (const float*)d_in[21 + 4 * l];
    }
    l1[0].CinEff = 18;  l1[0].Cpad = 32;
    l1[1].CinEff = 64;  l1[1].Cpad = 64;
    l1[2].CinEff = 64;  l1[2].Cpad = 64;
    for (int l = 0; l < 3; ++l) { l2[l].CinEff = 128; l2[l].Cpad = 128; }
    const float* scW1 = (const float*)d_in[15];
    const float* scG1 = (const float*)d_in[16];
    const float* scB1 = (const float*)d_in[17];
    const float* scW2 = (const float*)d_in[30];
    const float* scG2 = (const float*)d_in[31];
    const float* scB2 = (const float*)d_in[32];

    // ---- workspace carve ----
    char* w = (char*)d_ws;
    float*     gs    = (float*)(w + 0);          //   8 B (mean, invstd)
    float*     chsum = (float*)(w + 256);        // 512 B
    float*     chsq  = (float*)(w + 768);        // 512 B (contiguous with chsum)
    float*     scsum = (float*)(w + 1280);       // 512 B
    float*     scsq  = (float*)(w + 1792);       // 512 B
    _Float16*  Wh    = (_Float16*)(w + 4096);    // 128*128 f16 = 32 KB
    float*     Xn    = (float*)(w + 65536);      // 64*9*256 f32   = 576 KB
    float*     X2    = (float*)(w + 655360);     // 64*64*256 f32  = 4 MB
    int*       nnidx = (int*)(w + 4849664);      // 262144 i32     = 1 MB
    float*     scbuf = (float*)(w + 6291456);    // 128*16384 f32  = 8 MB
    _Float16*  bufA  = (_Float16*)(w + 16777216);// 262144*128 f16 = 64 MiB
    _Float16*  bufB  = (_Float16*)(w + 83886080);// 64 MiB
    float*     X3    = (float*)(w + 150994944);  // 64*128*256 f32 = 8 MB

    const long nTot = (long)PN_B * 9 * PN_N;     // 147456
    pn_gstats<<<1, 256, 0, stream>>>(X, nTot, gs);
    pn_ftrans<<<(int)((nTot + 255) / 256), 256, 0, stream>>>(X, Xn, gs, ftg, ftb, nTot);

    pn_run_block(Xn, 9,  64,  l1, scW1, scG1, scB1, X2,
                 nnidx, bufA, bufB, Wh, chsum, chsq, scsum, scsq, scbuf, stream);
    pn_run_block(X2, 64, 128, l2, scW2, scG2, scB2, X3,
                 nnidx, bufA, bufB, Wh, chsum, chsq, scsum, scsq, scbuf, stream);

    pn_finalmean<<<(PN_B * 128 + 255) / 256, 256, 0, stream>>>(X3, (float*)d_out, 128);
}
